// Model_40252433498735
// MI455X (gfx1250) — matrix operand order repaired and checked by reading, not compiled, not run on hardware
//
#include <hip/hip_runtime.h>
#include <hip/hip_bf16.h>

// ---------------------------------------------------------------------------
// Types for WMMA operands (gfx1250, wave32)
// ---------------------------------------------------------------------------
typedef _Float16 half_t;
typedef __attribute__((ext_vector_type(16))) _Float16 v16h;
typedef __attribute__((ext_vector_type(8)))  _Float16 v8h;
typedef __attribute__((ext_vector_type(4)))  _Float16 v4h;
typedef __attribute__((ext_vector_type(8)))  float    v8f;

union V16U { v16h v; v8h h[2]; };

// Load 16 contiguous halves (two 16B vector loads)
__device__ __forceinline__ v16h load_frag16(const half_t* p) {
    V16U u;
    u.h[0] = *(const v8h*)(p);
    u.h[1] = *(const v8h*)(p + 8);
    return u.v;
}

// B fragment: packed so lane `l` owns halves [l*16 .. l*16+16)
__device__ __forceinline__ v16h loadB(const half_t* frag, int lane) {
    return load_frag16(frag + lane * 16);
}

// A fragment (16x32 f16) from a row-major [., K] row for this lane's M row.
// Contiguous layout: lanes 0-15: K=kbase+0..15
//             lanes16-31: K=kbase+16..31  (khalf = 0 or 8)
__device__ __forceinline__ v16h loadA(const half_t* row, int kbase, int khalf) {
    V16U u;
    u.h[0] = *(const v8h*)(row + kbase + 2 * khalf);
    u.h[1] = *(const v8h*)(row + kbase + 2 * khalf + 8);
    return u.v;
}

__device__ __forceinline__ v8f wmma16(v16h a, v16h b, v8f c) {
    return __builtin_amdgcn_wmma_f32_16x16x32_f16(false, a, false, b,
                                                  (short)0, c, false, false);
}

__device__ __forceinline__ float sigmoidf_(float x) { return 1.0f / (1.0f + __expf(-x)); }
__device__ __forceinline__ float tanhf_(float x)    { return 2.0f / (1.0f + __expf(-2.0f * x)) - 1.0f; }

// ---------------------------------------------------------------------------
// Problem constants
// ---------------------------------------------------------------------------
#define N_NODES 30000
#define SEQ_LEN 32
#define VOCAB   10000
#define RNN_H   128
#define RGCN_IN 64
#define N_RELS  8
#define N_EDGES 480000

// ---------------------------------------------------------------------------
// Prep kernels: f32->f16 conversions and WMMA B-fragment packing
// ---------------------------------------------------------------------------
__global__ void f2h_kernel(const float* __restrict__ in, half_t* __restrict__ out, int n) {
    int i = blockIdx.x * blockDim.x + threadIdx.x;
    if (i < n) out[i] = (half_t)in[i];
}

__global__ void bias_add_kernel(const float* __restrict__ a, const float* __restrict__ b,
                                float* __restrict__ o, int n) {
    int i = blockIdx.x * blockDim.x + threadIdx.x;
    if (i < n) o[i] = a[i] + b[i];
}

// Pack concat(W_ih, W_hh) as B fragments: K=256 (x:0-127, h:128-255), N=512.
// Fragment order [kt(8)][nt(32)], 512 halves each; lane l owns 16 halves:
//   k = kt*32 + (l>=16 ? 16 : 0) + j,  n = nt*16 + (l&15)
__global__ void pack_lstm_w_kernel(const float* __restrict__ w_ih,
                                   const float* __restrict__ w_hh,
                                   half_t* __restrict__ dst) {
    int idx = blockIdx.x * blockDim.x + threadIdx.x;
    if (idx >= 8 * 32 * 512) return;
    int j  = idx & 15;
    int l  = (idx >> 4) & 31;
    int f  = idx >> 9;
    int nt = f & 31;
    int kt = f >> 5;
    int k = kt * 32 + ((l >> 4) << 4) + j;
    int n = nt * 16 + (l & 15);
    float v = (k < 128) ? w_ih[n * 128 + k] : w_hh[n * 128 + (k - 128)];
    dst[idx] = (half_t)v;
}

// fc_w [64,128] -> B fragments, [kt(4)][nt(4)]
__global__ void pack_fc_kernel(const float* __restrict__ fc_w, half_t* __restrict__ dst) {
    int idx = blockIdx.x * blockDim.x + threadIdx.x;
    if (idx >= 4 * 4 * 512) return;
    int j  = idx & 15;
    int l  = (idx >> 4) & 31;
    int f  = idx >> 9;
    int nt = f & 3;
    int kt = f >> 2;
    int k = kt * 32 + ((l >> 4) << 4) + j;
    int n = nt * 16 + (l & 15);
    dst[idx] = (half_t)fc_w[n * 128 + k];
}

// RGCN: W_r = sum_b wcomp[r,b]*basis[b]  (basis [4][64][OUT]); fragments
// ordered [r(8)][kt(2)][nt(OUT/16)]
__global__ void pack_rgcn_kernel(const float* __restrict__ basis,
                                 const float* __restrict__ wcomp,
                                 half_t* __restrict__ dst, int OUT) {
    int NT = OUT / 16;
    int total = 8 * 2 * NT * 512;
    int idx = blockIdx.x * blockDim.x + threadIdx.x;
    if (idx >= total) return;
    int j  = idx & 15;
    int l  = (idx >> 4) & 31;
    int f  = idx >> 9;
    int nt = f % NT;
    int kt = (f / NT) & 1;
    int r  = f / (NT * 2);
    int k = kt * 32 + ((l >> 4) << 4) + j;   // input-feature index (0..63)
    int n = nt * 16 + (l & 15);              // output index
    float v = 0.f;
#pragma unroll
    for (int b = 0; b < 4; ++b)
        v += wcomp[r * 4 + b] * basis[(b * 64 + k) * OUT + n];
    dst[idx] = (half_t)v;
}

// ---------------------------------------------------------------------------
// Fused reverse-LSTM (32 steps) + FC(128->64).
// 8 waves/block: wave = 16-wide hidden tile; block owns 16 nodes.
// Per wave per step: 32 x v_wmma_f32_16x16x32_f16 (4 gates x K=256).
// Cell state lives in VGPRs across the whole sequence; h exchanged via LDS f16.
// Token gather for step t+1 is software-pipelined behind step t's WMMAs, and
// all A fragments of a step are batch-loaded so the compiler can emit one
// load clause + single wait before the 32-WMMA burst.
// ---------------------------------------------------------------------------
__global__ __launch_bounds__(256, 1)
void lstm_fc_kernel(const int* __restrict__ tokens,      // [N, 32]
                    const half_t* __restrict__ emb_h,    // [VOCAB, 128] f16
                    const half_t* __restrict__ wfrag,    // packed concat weights
                    const float* __restrict__ bias,      // b_ih+b_hh [512]
                    const half_t* __restrict__ fcfrag,   // packed fc weights
                    const float* __restrict__ fc_b,      // [64]
                    float* __restrict__ feats)           // [N, 64]
{
    __shared__ __align__(16) half_t h_lds[16 * 128];
    const int node_base = blockIdx.x * 16;
    const int tid   = threadIdx.x;
    const int lane  = tid & 31;
    const int wave  = tid >> 5;              // hidden tile 0..7
    const int mrow  = lane & 15;             // A-matrix / C-matrix column mapping
    const int khalf = (lane >> 4) << 3;      // 0 or 8 (A-fragment half select)

    for (int i = tid; i < 16 * 128; i += 256) h_lds[i] = (half_t)0.f;
    __syncthreads();

    const float bi = bias[0 * 128 + wave * 16 + mrow];
    const float bf = bias[1 * 128 + wave * 16 + mrow];
    const float bg = bias[2 * 128 + wave * 16 + mrow];
    const float bo = bias[3 * 128 + wave * 16 + mrow];

    v8f cst;
#pragma unroll
    for (int j = 0; j < 8; ++j) cst[j] = 0.f;

    const int tok_base = (node_base + mrow) * SEQ_LEN;
    int tok = tokens[tok_base + (SEQ_LEN - 1)];           // first (reverse) token

    for (int t = 0; t < SEQ_LEN; ++t) {
        const half_t* xrow = emb_h + (size_t)tok * 128;

        // Pipeline: kick off next step's token gather + prefetch its row.
        int ntok = (t < SEQ_LEN - 1) ? tokens[tok_base + (SEQ_LEN - 2 - t)] : tok;
        __builtin_prefetch(emb_h + (size_t)ntok * 128, 0, 3);

        // Batch-load all A fragments for this step (8 global + 8 LDS b128s).
        v16h ax[4], ah[4];
#pragma unroll
        for (int kt = 0; kt < 4; ++kt) ax[kt] = loadA(xrow, kt * 32, khalf);
#pragma unroll
        for (int kt = 0; kt < 4; ++kt) ah[kt] = loadA(h_lds + mrow * 128, kt * 32, khalf);

        v8f zi, zf, zg, zo;
#pragma unroll
        for (int j = 0; j < 8; ++j) { zi[j] = bi; zf[j] = bf; zg[j] = bg; zo[j] = bo; }

        // x @ W_ih^T  (K tiles 0..3)
#pragma unroll
        for (int kt = 0; kt < 4; ++kt) {
            zi = wmma16(ax[kt], loadB(wfrag + (size_t)(kt * 32 + 0 * 8 + wave) * 512, lane), zi);
            zf = wmma16(ax[kt], loadB(wfrag + (size_t)(kt * 32 + 1 * 8 + wave) * 512, lane), zf);
            zg = wmma16(ax[kt], loadB(wfrag + (size_t)(kt * 32 + 2 * 8 + wave) * 512, lane), zg);
            zo = wmma16(ax[kt], loadB(wfrag + (size_t)(kt * 32 + 3 * 8 + wave) * 512, lane), zo);
        }
        // h @ W_hh^T  (K tiles 4..7)
#pragma unroll
        for (int kt = 0; kt < 4; ++kt) {
            zi = wmma16(ah[kt], loadB(wfrag + (size_t)((4 + kt) * 32 + 0 * 8 + wave) * 512, lane), zi);
            zf = wmma16(ah[kt], loadB(wfrag + (size_t)((4 + kt) * 32 + 1 * 8 + wave) * 512, lane), zf);
            zg = wmma16(ah[kt], loadB(wfrag + (size_t)((4 + kt) * 32 + 2 * 8 + wave) * 512, lane), zg);
            zo = wmma16(ah[kt], loadB(wfrag + (size_t)((4 + kt) * 32 + 3 * 8 + wave) * 512, lane), zo);
        }

        v8f hnew;
#pragma unroll
        for (int j = 0; j < 8; ++j) {
            float c = sigmoidf_(zf[j]) * cst[j] + sigmoidf_(zi[j]) * tanhf_(zg[j]);
            cst[j]  = c;
            hnew[j] = sigmoidf_(zo[j]) * tanhf_(c);
        }

        __syncthreads();   // all waves done reading previous h
#pragma unroll
        for (int j = 0; j < 8; ++j) {
            int m = j + ((lane >> 4) << 3);   // C-layout: VGPR j -> row j / j+8
            h_lds[m * 128 + wave * 16 + mrow] = (half_t)hnew[j];
        }
        __syncthreads();   // new h visible to all waves

        tok = ntok;
    }

    // FC: feats = h_last @ fc_w^T + fc_b  (waves 0..3, one 16-wide out tile each)
    if (wave < 4) {
        const int nt = wave;
        v8f acc;
        const float fb = fc_b[nt * 16 + mrow];
#pragma unroll
        for (int j = 0; j < 8; ++j) acc[j] = fb;
#pragma unroll
        for (int kt = 0; kt < 4; ++kt) {
            v16h a = loadA(h_lds + mrow * 128, kt * 32, khalf);
            acc = wmma16(a, loadB(fcfrag + (size_t)(kt * 4 + nt) * 512, lane), acc);
        }
#pragma unroll
        for (int j = 0; j < 8; ++j) {
            int m = j + ((lane >> 4) << 3);
            feats[(size_t)(node_base + m) * 64 + nt * 16 + mrow] = acc[j];
        }
    }
}

// ---------------------------------------------------------------------------
// RGCN message scatter: agg[(dst*8+rel)*64 + f] += h[src][f] * norm
// 4 threads per edge (16 feats each) for more atomic-pipe parallelism.
// ---------------------------------------------------------------------------
__global__ void scatter_kernel(const float* __restrict__ feats,
                               const int* __restrict__ src,
                               const int* __restrict__ dst,
                               const int* __restrict__ rel,
                               const float* __restrict__ enorm,
                               float* __restrict__ agg, int E) {
    int tid = blockIdx.x * blockDim.x + threadIdx.x;
    int e = tid >> 2;
    int q0 = (tid & 3) * 4;                  // float4 quarter index
    if (e >= E) return;
    const float4* s = (const float4*)(feats + (size_t)src[e] * 64);
    const float nm = enorm[e];
    float* a = agg + ((size_t)dst[e] * N_RELS + rel[e]) * 64;
#pragma unroll
    for (int q = q0; q < q0 + 4; ++q) {
        float4 v = s[q];
        atomicAdd(a + q * 4 + 0, v.x * nm);
        atomicAdd(a + q * 4 + 1, v.y * nm);
        atomicAdd(a + q * 4 + 2, v.z * nm);
        atomicAdd(a + q * 4 + 3, v.w * nm);
    }
}

// ---------------------------------------------------------------------------
// RGCN transform: out[n] = act( sum_r agg[n,r,:] @ W_r ).
// Block = 16 nodes; NT = NOUT/16 waves, each computes one 16-wide out tile
// with 16 WMMAs (8 rel x 2 K-tiles). agg staged to LDS as f16 (vectorized:
// float4 in -> v4h 8-byte LDS store).
// ---------------------------------------------------------------------------
template <int NOUT, bool RELU>
__global__ void rgcn_transform_kernel(const float* __restrict__ agg,
                                      const half_t* __restrict__ wfrag,
                                      float* __restrict__ out) {
    constexpr int NT = NOUT / 16;
    __shared__ __align__(16) half_t a_lds[16 * 512];
    const int node_base = blockIdx.x * 16;
    const int tid  = threadIdx.x;
    const int lane = tid & 31;
    const int wv   = tid >> 5;
    const float4* ag4 = (const float4*)(agg + (size_t)node_base * 512);
    for (int i = tid; i < 16 * 512 / 4; i += NT * 32) {
        float4 v = ag4[i];
        v4h h; h[0] = (half_t)v.x; h[1] = (half_t)v.y; h[2] = (half_t)v.z; h[3] = (half_t)v.w;
        *(v4h*)(a_lds + i * 4) = h;
    }
    __syncthreads();

    const int mrow  = lane & 15;
    const int khalf = (lane >> 4) << 3;
    v8f acc;
#pragma unroll
    for (int j = 0; j < 8; ++j) acc[j] = 0.f;

#pragma unroll
    for (int r = 0; r < N_RELS; ++r) {
#pragma unroll
        for (int kt = 0; kt < 2; ++kt) {
            v16h a = loadA(a_lds + mrow * 512 + r * 64, kt * 32, khalf);
            v16h b = loadB(wfrag + (size_t)((r * 2 + kt) * NT + wv) * 512, lane);
            acc = wmma16(a, b, acc);
        }
    }
#pragma unroll
    for (int j = 0; j < 8; ++j) {
        int m = j + ((lane >> 4) << 3);
        float x = acc[j];
        if (RELU) x = fmaxf(x, 0.f);
        out[(size_t)(node_base + m) * NOUT + wv * 16 + mrow] = x;
    }
}

// ---------------------------------------------------------------------------
// Launch
// ---------------------------------------------------------------------------
extern "C" void kernel_launch(void* const* d_in, const int* in_sizes, int n_in,
                              void* d_out, int out_size, void* d_ws, size_t ws_size,
                              hipStream_t stream) {
    (void)in_sizes; (void)n_in; (void)out_size; (void)ws_size;
    const int*   tokens = (const int*)d_in[0];
    // d_in[1] = sequence_length (unused by the reference math)
    const int*   src    = (const int*)d_in[2];
    const int*   dst    = (const int*)d_in[3];
    const int*   rel    = (const int*)d_in[4];
    const float* enorm  = (const float*)d_in[5];
    const float* emb    = (const float*)d_in[6];
    const float* w_ih   = (const float*)d_in[7];
    const float* w_hh   = (const float*)d_in[8];
    const float* b_ih   = (const float*)d_in[9];
    const float* b_hh   = (const float*)d_in[10];
    const float* fc_w   = (const float*)d_in[11];
    const float* fc_b   = (const float*)d_in[12];
    const float* basis0 = (const float*)d_in[13];
    const float* wcomp0 = (const float*)d_in[14];
    const float* basis1 = (const float*)d_in[15];
    const float* wcomp1 = (const float*)d_in[16];
    const float* basis2 = (const float*)d_in[17];
    const float* wcomp2 = (const float*)d_in[18];
    float* out = (float*)d_out;

    char* ws = (char*)d_ws;
    size_t off = 0;
    auto walloc = [&](size_t bytes) -> void* {
        void* p = ws + off;
        off = (off + bytes + 255) & ~(size_t)255;
        return p;
    };
    half_t* emb_h = (half_t*)walloc((size_t)VOCAB * 128 * 2);
    half_t* wcat  = (half_t*)walloc((size_t)8 * 32 * 512 * 2);
    half_t* fcfr  = (half_t*)walloc((size_t)4 * 4 * 512 * 2);
    half_t* w0fr  = (half_t*)walloc((size_t)8 * 2 * 4 * 512 * 2);
    half_t* w1fr  = (half_t*)walloc((size_t)8 * 2 * 4 * 512 * 2);
    half_t* w2fr  = (half_t*)walloc((size_t)8 * 2 * 1 * 512 * 2);
    float*  bias  = (float*)walloc(512 * 4);
    float*  fA    = (float*)walloc((size_t)N_NODES * 64 * 4);
    float*  fB    = (float*)walloc((size_t)N_NODES * 64 * 4);
    float*  agg   = (float*)walloc((size_t)N_NODES * N_RELS * 64 * 4);

    const int nEmb = VOCAB * 128;
    f2h_kernel<<<(nEmb + 255) / 256, 256, 0, stream>>>(emb, emb_h, nEmb);
    pack_lstm_w_kernel<<<(8 * 32 * 512 + 255) / 256, 256, 0, stream>>>(w_ih, w_hh, wcat);
    pack_fc_kernel<<<(4 * 4 * 512 + 255) / 256, 256, 0, stream>>>(fc_w, fcfr);
    bias_add_kernel<<<2, 256, 0, stream>>>(b_ih, b_hh, bias, 512);
    pack_rgcn_kernel<<<(8 * 2 * 4 * 512 + 255) / 256, 256, 0, stream>>>(basis0, wcomp0, w0fr, 64);
    pack_rgcn_kernel<<<(8 * 2 * 4 * 512 + 255) / 256, 256, 0, stream>>>(basis1, wcomp1, w1fr, 64);
    pack_rgcn_kernel<<<(8 * 2 * 1 * 512 + 255) / 256, 256, 0, stream>>>(basis2, wcomp2, w2fr, 16);

    lstm_fc_kernel<<<N_NODES / 16, 256, 0, stream>>>(tokens, emb_h, wcat, bias, fcfr, fc_b, fA);

    const size_t aggBytes = (size_t)N_NODES * N_RELS * 64 * 4;
    const int eBlocks = (N_EDGES * 4 + 255) / 256;

    // layer 1: fA -> fB (relu)
    hipMemsetAsync(agg, 0, aggBytes, stream);
    scatter_kernel<<<eBlocks, 256, 0, stream>>>(fA, src, dst, rel, enorm, agg, N_EDGES);
    rgcn_transform_kernel<64, true><<<N_NODES / 16, 128, 0, stream>>>(agg, w0fr, fB);

    // layer 2: fB -> fA (relu)
    hipMemsetAsync(agg, 0, aggBytes, stream);
    scatter_kernel<<<eBlocks, 256, 0, stream>>>(fB, src, dst, rel, enorm, agg, N_EDGES);
    rgcn_transform_kernel<64, true><<<N_NODES / 16, 128, 0, stream>>>(agg, w1fr, fA);

    // layer 3: fA -> out (no activation, 16 classes)
    hipMemsetAsync(agg, 0, aggBytes, stream);
    scatter_kernel<<<eBlocks, 256, 0, stream>>>(fA, src, dst, rel, enorm, agg, N_EDGES);
    rgcn_transform_kernel<16, false><<<N_NODES / 16, 32, 0, stream>>>(agg, w2fr, out);
}
